// GraphSAGEClassifier_33346126086715
// MI455X (gfx1250) — matrix operand order repaired and checked by reading, not compiled, not run on hardware
//
#include <hip/hip_runtime.h>
#include <math.h>

typedef _Float16 v16h __attribute__((ext_vector_type(16)));
typedef float    v8f  __attribute__((ext_vector_type(8)));

#define NNODES 100000
#define NEDGES 1600000
#define FDIM   128
#define HDIM   64
#define NGRAPH 128
#define NCLS   10
#define BN_EPS 1e-5f

__device__ __forceinline__ void atomic_add_f32(float* p, float v) {
  __hip_atomic_fetch_add(p, v, __ATOMIC_RELAXED, __HIP_MEMORY_SCOPE_AGENT);
}

__global__ void zero_f32_kernel(float* __restrict__ p, int n) {
  int t = blockIdx.x * blockDim.x + threadIdx.x;
  if (t < n) p[t] = 0.0f;
}

__global__ void degree_kernel(const int* __restrict__ dst, float* __restrict__ cnt, int ne) {
  int e = blockIdx.x * blockDim.x + threadIdx.x;
  if (e < ne) atomic_add_f32(&cnt[dst[e]], 1.0f);
}

__global__ void graph_count_kernel(const int* __restrict__ batch, float* __restrict__ gcnt, int n) {
  int i = blockIdx.x * blockDim.x + threadIdx.x;
  if (i < n) atomic_add_f32(&gcnt[batch[i]], 1.0f);
}

// Fused dual GEMM: Y = X @ Wl, R = X @ Wr.  X:[nrows,K] f32, W:[K,64] f32 row-major.
// One wave computes a 16-row x 64-col tile of BOTH outputs via v_wmma_f32_16x16x32_f16.
// Weights are staged into LDS pre-swizzled into the exact B-fragment lane layout, so the
// inner loop fetches each v16h fragment with contiguous 32B LDS reads (ds_load_b128 x2).
template <int K>
__global__ __launch_bounds__(256) void sage_dual_matmul_kernel(
    const float* __restrict__ X, const float* __restrict__ Wl, const float* __restrict__ Wr,
    float* __restrict__ Y, float* __restrict__ R, int nrows) {
  // layout: [K/32 ktiles][4 ntiles][32 lanes][16 halves]
  __shared__ __align__(32) _Float16 sWl[(K / 32) * 4 * 32 * 16];
  __shared__ __align__(32) _Float16 sWr[(K / 32) * 4 * 32 * 16];
  for (int idx = threadIdx.x; idx < K * HDIM; idx += blockDim.x) {
    int k = idx >> 6;                 // W row   (HDIM == 64)
    int n = idx & 63;                 // W col
    int kt = k >> 5;                  // 32-wide k-tile
    int hf = (k >> 3) & 1;            // lane half select
    int j  = (k & 7) | (((k >> 4) & 1) << 3); // element within fragment
    int tt = n >> 4;                  // 16-wide n-tile
    int ln = hf * 16 + (n & 15);      // owning lane
    int d  = (((kt << 2) + tt) * 32 + ln) * 16 + j;
    sWl[d] = (_Float16)Wl[idx];
    sWr[d] = (_Float16)Wr[idx];
  }
  __syncthreads();

  int wave = (int)((blockIdx.x * blockDim.x + threadIdx.x) >> 5);
  int lane = (int)(threadIdx.x & 31);
  int m0 = wave * 16;
  if (m0 >= nrows) return;            // wave-uniform: nrows is a multiple of 16

  int mrow = m0 + (lane & 15);
  int hi   = lane >> 4;               // lane half: 0 or 1
  int ncol = lane & 15;

  v8f zero8 = {0.f, 0.f, 0.f, 0.f, 0.f, 0.f, 0.f, 0.f};
  v8f accY[4] = {zero8, zero8, zero8, zero8};
  v8f accR[4] = {zero8, zero8, zero8, zero8};

  #pragma unroll
  for (int k0 = 0; k0 < K; k0 += 32) {
    // A fragment (16x32 f16): lanes 0-15 -> K {0..7,16..23}, lanes 16-31 -> K {8..15,24..31}
    const float4* ap = (const float4*)(X + (size_t)mrow * K + k0 + hi * 8);
    float4 x0 = ap[0], x1 = ap[1];    // k .. k+7
    float4 x2 = ap[4], x3 = ap[5];    // k+16 .. k+23
    v16h a;
    a[0]  = (_Float16)x0.x; a[1]  = (_Float16)x0.y; a[2]  = (_Float16)x0.z; a[3]  = (_Float16)x0.w;
    a[4]  = (_Float16)x1.x; a[5]  = (_Float16)x1.y; a[6]  = (_Float16)x1.z; a[7]  = (_Float16)x1.w;
    a[8]  = (_Float16)x2.x; a[9]  = (_Float16)x2.y; a[10] = (_Float16)x2.z; a[11] = (_Float16)x2.w;
    a[12] = (_Float16)x3.x; a[13] = (_Float16)x3.y; a[14] = (_Float16)x3.z; a[15] = (_Float16)x3.w;

    int base = (k0 >> 5) << 2;        // kt*4
    #pragma unroll
    for (int t = 0; t < 4; ++t) {
      v16h bl = *(const v16h*)&sWl[((base + t) * 32 + lane) * 16];
      v16h br = *(const v16h*)&sWr[((base + t) * 32 + lane) * 16];
      accY[t] = __builtin_amdgcn_wmma_f32_16x16x32_f16(false, a, false, bl, (short)0, accY[t], false, false);
      accR[t] = __builtin_amdgcn_wmma_f32_16x16x32_f16(false, a, false, br, (short)0, accR[t], false, false);
    }
  }

  // C/D layout: VGPR r -> M = r (+8 for lanes 16-31), N = lane&15
  #pragma unroll
  for (int t = 0; t < 4; ++t) {
    #pragma unroll
    for (int r = 0; r < 8; ++r) {
      int row = m0 + hi * 8 + r;
      int col = t * 16 + ncol;
      Y[(size_t)row * HDIM + col] = accY[t][r];
      R[(size_t)row * HDIM + col] = accR[t][r];
    }
  }
}

// Edge scatter: agg[dst] += y[src], one thread per (edge, 4-feature chunk).
__global__ void scatter_sum_kernel(const float* __restrict__ y, const int* __restrict__ src,
                                   const int* __restrict__ dst, float* __restrict__ agg, int ne) {
  int t = blockIdx.x * blockDim.x + threadIdx.x;
  int e = t >> 4;
  if (e >= ne) return;
  int q = (t & 15) << 2;              // feature base: 0,4,...,60
  int s = src[e], d = dst[e];
  float4 v = *(const float4*)(y + s * HDIM + q);
  float* ag = agg + d * HDIM + q;
  atomic_add_f32(ag + 0, v.x);
  atomic_add_f32(ag + 1, v.y);
  atomic_add_f32(ag + 2, v.z);
  atomic_add_f32(ag + 3, v.w);
}

// h = relu(BN(agg/max(cnt,1) + bias + r)) ; written in place into agg buffer.
__global__ void finish_layer_kernel(float* hs, const float* __restrict__ rr,
                                    const float* __restrict__ cnt, const float* __restrict__ bias,
                                    const float* __restrict__ bg, const float* __restrict__ bb,
                                    const float* __restrict__ bm, const float* __restrict__ bv,
                                    int n) {
  int t = blockIdx.x * blockDim.x + threadIdx.x;
  if (t >= n * HDIM) return;
  int node = t >> 6;
  int f = t & 63;
  float mean = hs[t] / fmaxf(cnt[node], 1.0f);
  float v = mean + bias[f] + rr[t];
  v = (v - bm[f]) * rsqrtf(bv[f] + BN_EPS) * bg[f] + bb[f];
  hs[t] = fmaxf(v, 0.0f);
}

__global__ void pool_kernel(const float* __restrict__ h, const int* __restrict__ batch,
                            float* __restrict__ gsum, int n) {
  int t = blockIdx.x * blockDim.x + threadIdx.x;
  if (t >= n * 16) return;
  int node = t >> 4;
  int q = (t & 15) << 2;
  int g = batch[node];
  float4 v = *(const float4*)(h + node * HDIM + q);
  float* gs = gsum + g * HDIM + q;
  atomic_add_f32(gs + 0, v.x);
  atomic_add_f32(gs + 1, v.y);
  atomic_add_f32(gs + 2, v.z);
  atomic_add_f32(gs + 3, v.w);
}

// Head: g = gsum/gcnt ; relu(BN3(g@Wc1+bc1)) @ Wc2 + bc2 ; log_softmax. 128 threads, 1 block.
__global__ void head_kernel(const float* __restrict__ gsum, const float* __restrict__ gcnt,
                            const float* __restrict__ Wc1, const float* __restrict__ bc1,
                            const float* __restrict__ g3, const float* __restrict__ b3,
                            const float* __restrict__ m3, const float* __restrict__ v3,
                            const float* __restrict__ Wc2, const float* __restrict__ bc2,
                            float* __restrict__ out) {
  int gi = threadIdx.x;
  if (gi >= NGRAPH) return;
  float vec[HDIM];
  float inv = 1.0f / fmaxf(gcnt[gi], 1.0f);
  #pragma unroll
  for (int k = 0; k < HDIM; ++k) vec[k] = gsum[gi * HDIM + k] * inv;
  float logit[NCLS];
  #pragma unroll
  for (int c = 0; c < NCLS; ++c) logit[c] = bc2[c];
  for (int hh = 0; hh < HDIM; ++hh) {
    float s = bc1[hh];
    #pragma unroll
    for (int k = 0; k < HDIM; ++k) s += vec[k] * Wc1[k * HDIM + hh];
    s = (s - m3[hh]) * rsqrtf(v3[hh] + BN_EPS) * g3[hh] + b3[hh];
    s = fmaxf(s, 0.0f);
    #pragma unroll
    for (int c = 0; c < NCLS; ++c) logit[c] += s * Wc2[hh * NCLS + c];
  }
  float mx = logit[0];
  #pragma unroll
  for (int c = 1; c < NCLS; ++c) mx = fmaxf(mx, logit[c]);
  float se = 0.0f;
  #pragma unroll
  for (int c = 0; c < NCLS; ++c) se += expf(logit[c] - mx);
  float lse = logf(se) + mx;
  #pragma unroll
  for (int c = 0; c < NCLS; ++c) out[gi * NCLS + c] = logit[c] - lse;
}

extern "C" void kernel_launch(void* const* d_in, const int* in_sizes, int n_in,
                              void* d_out, int out_size, void* d_ws, size_t ws_size,
                              hipStream_t stream) {
  (void)in_sizes; (void)n_in; (void)out_size; (void)ws_size;

  const float* x     = (const float*)d_in[0];
  const int*   ei    = (const int*)d_in[1];
  const int*   batch = (const int*)d_in[2];
  const float* W1l = (const float*)d_in[3];
  const float* W1r = (const float*)d_in[4];
  const float* b1  = (const float*)d_in[5];
  const float* W2l = (const float*)d_in[6];
  const float* W2r = (const float*)d_in[7];
  const float* b2  = (const float*)d_in[8];
  const float* bn1g = (const float*)d_in[9],  *bn1b = (const float*)d_in[10];
  const float* bn1m = (const float*)d_in[11], *bn1v = (const float*)d_in[12];
  const float* bn2g = (const float*)d_in[13], *bn2b = (const float*)d_in[14];
  const float* bn2m = (const float*)d_in[15], *bn2v = (const float*)d_in[16];
  const float* bn3g = (const float*)d_in[17], *bn3b = (const float*)d_in[18];
  const float* bn3m = (const float*)d_in[19], *bn3v = (const float*)d_in[20];
  const float* Wc1 = (const float*)d_in[21];
  const float* bc1 = (const float*)d_in[22];
  const float* Wc2 = (const float*)d_in[23];
  const float* bc2 = (const float*)d_in[24];

  const int* src = ei;            // edge_index[0, :]
  const int* dst = ei + NEDGES;   // edge_index[1, :]

  // workspace layout (floats)
  const size_t N64 = (size_t)NNODES * HDIM;
  float* ws   = (float*)d_ws;
  float* ybuf = ws;                 // [N,64] aggregated-branch transform
  float* rbuf = ybuf + N64;         // [N,64] root-branch transform
  float* abuf = rbuf + N64;         // [N,64] edge-sum / h (in place)
  float* cnt  = abuf + N64;         // [N]
  float* gcnt = cnt + NNODES;       // [G]
  float* gsum = gcnt + NGRAPH;      // [G,64]

  const int TPB = 256;
  auto nblk = [](long long n) { return (unsigned)((n + 255) / 256); };

  const int nNH = NNODES * HDIM;         // 6.4M
  const int nGH = NGRAPH * HDIM;

  // ---- init accumulators ----
  zero_f32_kernel<<<nblk(NNODES), TPB, 0, stream>>>(cnt, NNODES);
  zero_f32_kernel<<<1, TPB, 0, stream>>>(gcnt, NGRAPH);
  zero_f32_kernel<<<nblk(nGH), TPB, 0, stream>>>(gsum, nGH);
  zero_f32_kernel<<<nblk(nNH), TPB, 0, stream>>>(abuf, nNH);

  // ---- degrees / graph sizes ----
  degree_kernel<<<nblk(NEDGES), TPB, 0, stream>>>(dst, cnt, NEDGES);
  graph_count_kernel<<<nblk(NNODES), TPB, 0, stream>>>(batch, gcnt, NNODES);

  // ---- layer 1: transform (WMMA), aggregate, epilogue ----
  const unsigned mmBlocks = nblk((long long)(NNODES / 16) * 32); // 8 waves/block, 1 tile/wave
  sage_dual_matmul_kernel<FDIM><<<mmBlocks, TPB, 0, stream>>>(x, W1l, W1r, ybuf, rbuf, NNODES);
  scatter_sum_kernel<<<nblk((long long)NEDGES * 16), TPB, 0, stream>>>(ybuf, src, dst, abuf, NEDGES);
  finish_layer_kernel<<<nblk(nNH), TPB, 0, stream>>>(abuf, rbuf, cnt, b1,
                                                     bn1g, bn1b, bn1m, bn1v, NNODES);
  // abuf now holds h1

  // ---- layer 2 ----
  sage_dual_matmul_kernel<HDIM><<<mmBlocks, TPB, 0, stream>>>(abuf, W2l, W2r, ybuf, rbuf, NNODES);
  zero_f32_kernel<<<nblk(nNH), TPB, 0, stream>>>(abuf, nNH);   // h1 dead; reuse as agg2
  scatter_sum_kernel<<<nblk((long long)NEDGES * 16), TPB, 0, stream>>>(ybuf, src, dst, abuf, NEDGES);
  finish_layer_kernel<<<nblk(nNH), TPB, 0, stream>>>(abuf, rbuf, cnt, b2,
                                                     bn2g, bn2b, bn2m, bn2v, NNODES);
  // abuf now holds h2

  // ---- global mean pool + head ----
  pool_kernel<<<nblk((long long)NNODES * 16), TPB, 0, stream>>>(abuf, batch, gsum, NNODES);
  head_kernel<<<1, NGRAPH, 0, stream>>>(gsum, gcnt, Wc1, bc1, bn3g, bn3b, bn3m, bn3v,
                                        Wc2, bc2, (float*)d_out);
}
